// BitLinear_20349555048669
// MI455X (gfx1250) — matrix operand order repaired and checked by reading, not compiled, not run on hardware
//
#include <hip/hip_runtime.h>
#include <hip/hip_bf16.h>
#include <stdint.h>

typedef __attribute__((ext_vector_type(16))) _Float16 v16h;
typedef __attribute__((ext_vector_type(8)))  _Float16 v8h;
typedef __attribute__((ext_vector_type(4)))  _Float16 v4h;
typedef __attribute__((ext_vector_type(8)))  float    v8f;

#define BM 128
#define BN 128
#define BK 64
#define LDT 72                 // padded LDS row stride in halves (144 B, 16 B-aligned chunks)
#define TILE_H (BM * LDT)      // halves per LDS tile buffer

// ---------------------------------------------------------------------------
// Pass 1: sum(|w|) reduction (wave32 shuffle reduce + one atomic per wave)
// ---------------------------------------------------------------------------
__global__ __launch_bounds__(256) void absum_kernel(const float* __restrict__ w,
                                                    float* __restrict__ sum_out,
                                                    int n4) {
  const float4* w4 = (const float4*)w;
  float s = 0.f;
  for (long i = (long)blockIdx.x * 256 + threadIdx.x; i < n4;
       i += (long)gridDim.x * 256) {
    float4 v = w4[i];
    s += fabsf(v.x) + fabsf(v.y) + fabsf(v.z) + fabsf(v.w);
  }
  for (int off = 16; off > 0; off >>= 1) s += __shfl_down(s, off, 32);
  if ((threadIdx.x & 31) == 0) atomicAdd(sum_out, s);
}

__device__ __forceinline__ float ternarize(float v, float thr) {
  return (fabsf(v) >= thr) ? (v >= 0.f ? 1.f : -1.f) : 0.f;
}

// ---------------------------------------------------------------------------
// Pass 2: W fp32 -> ternary f16 (exact representation of {-1,0,1})
// ---------------------------------------------------------------------------
__global__ __launch_bounds__(256) void quant_w_kernel(const float* __restrict__ w,
                                                      _Float16* __restrict__ wq,
                                                      const float* __restrict__ sum_abs,
                                                      int n4, float inv_n) {
  const float thr = 0.7f * (*sum_abs) * inv_n;  // 0.7 * mean(|w|)
  const float4* w4 = (const float4*)w;
  v4h* q4 = (v4h*)wq;
  for (long i = (long)blockIdx.x * 256 + threadIdx.x; i < n4;
       i += (long)gridDim.x * 256) {
    float4 v = w4[i];
    v4h q;
    q[0] = (_Float16)ternarize(v.x, thr);
    q[1] = (_Float16)ternarize(v.y, thr);
    q[2] = (_Float16)ternarize(v.z, thr);
    q[3] = (_Float16)ternarize(v.w, thr);
    q4[i] = q;
  }
}

// ---------------------------------------------------------------------------
// Pass 3: X fp32 -> f16
// ---------------------------------------------------------------------------
__global__ __launch_bounds__(256) void convert_x_kernel(const float* __restrict__ x,
                                                        _Float16* __restrict__ xh,
                                                        int n4) {
  const float4* x4 = (const float4*)x;
  v4h* h4 = (v4h*)xh;
  for (long i = (long)blockIdx.x * 256 + threadIdx.x; i < n4;
       i += (long)gridDim.x * 256) {
    float4 v = x4[i];
    v4h h;
    h[0] = (_Float16)v.x;
    h[1] = (_Float16)v.y;
    h[2] = (_Float16)v.z;
    h[3] = (_Float16)v.w;
    h4[i] = h;
  }
}

// ---------------------------------------------------------------------------
// Pass 4: Y[M,N] = A[M,K] * B[N,K]^T + bias  via v_wmma_f32_16x16x32_f16
//   - 128x128x64 block tile, 8 waves, each wave 2x4 grid of 16x16 WMMA tiles
//   - global -> LDS via async DMA (GLOBAL_LOAD_ASYNC_TO_LDS_B128, ASYNCcnt)
//   - double-buffered LDS; last iteration peeled so the hot loop is branch-free
// ---------------------------------------------------------------------------
__global__ __launch_bounds__(256) void gemm_f16_wmma_async_kernel(
    const _Float16* __restrict__ A,   // [M,K] f16 activations
    const _Float16* __restrict__ B,   // [N,K] f16 ternary weights
    const float*    __restrict__ bias,
    float*          __restrict__ C,   // [M,N] fp32
    int M, int N, int K) {
  __shared__ _Float16 sA[2 * TILE_H];
  __shared__ _Float16 sB[2 * TILE_H];

  const int tid   = threadIdx.x;
  const int lane  = tid & 31;
  const int wave  = tid >> 5;
  const int waveM = (wave & 3) * 32;  // 4 waves along M
  const int waveN = (wave >> 2) * 64; // 2 waves along N
  const int lrow  = lane & 15;
  const int lgrp  = lane >> 4;

  const int block_m = blockIdx.y * BM;
  const int block_n = blockIdx.x * BN;

  v8f acc[2][4];
#pragma unroll
  for (int mt = 0; mt < 2; ++mt)
#pragma unroll
    for (int nt = 0; nt < 4; ++nt)
#pragma unroll
      for (int j = 0; j < 8; ++j) acc[mt][nt][j] = 0.f;

  // Tile = 128 rows x 64 halves = 1024 x 16B chunks; 4 A + 4 B chunks/thread.
  // LDS byte offsets for the async DMA come from truncating the generic
  // pointer: flat-aperture low 32 bits are the wave-relative LDS offset.
  const _Float16* gA[4];
  const _Float16* gB[4];
  uint32_t lA[4], lB[4];
  const uint32_t ldsA0 = (uint32_t)(uintptr_t)&sA[0];
  const uint32_t ldsB0 = (uint32_t)(uintptr_t)&sB[0];
#pragma unroll
  for (int i = 0; i < 4; ++i) {
    const int c   = tid + 256 * i;
    const int row = c >> 3;        // 0..127
    const int kc  = (c & 7) * 8;   // halves within row
    gA[i] = A + (size_t)(block_m + row) * K + kc;
    gB[i] = B + (size_t)(block_n + row) * K + kc;
    lA[i] = ldsA0 + (uint32_t)(row * LDT + kc) * 2;
    lB[i] = ldsB0 + (uint32_t)(row * LDT + kc) * 2;
  }

  auto issue_tile = [&](int kt, uint32_t boff) {
#pragma unroll
    for (int i = 0; i < 4; ++i) {
      asm volatile("global_load_async_to_lds_b128 %0, %1, off"
                   :: "v"(lA[i] + boff), "v"(gA[i] + kt) : "memory");
      asm volatile("global_load_async_to_lds_b128 %0, %1, off"
                   :: "v"(lB[i] + boff), "v"(gB[i] + kt) : "memory");
    }
  };

  auto compute_tile = [&](const _Float16* bufA, const _Float16* bufB) {
#pragma unroll
    for (int kk = 0; kk < BK; kk += 32) {
      // A frag (16x32 f16): lanes 0-15 K {0..7,16..23}, lanes 16-31 K {8..15,24..31}
      v16h af[2];
#pragma unroll
      for (int mt = 0; mt < 2; ++mt) {
        const int row = waveM + mt * 16 + lrow;
        v8h lo = *(const v8h*)&bufA[row * LDT + kk + lgrp * 8];
        v8h hi = *(const v8h*)&bufA[row * LDT + kk + 16 + lgrp * 8];
        af[mt] = __builtin_shufflevector(lo, hi, 0, 1, 2, 3, 4, 5, 6, 7,
                                         8, 9, 10, 11, 12, 13, 14, 15);
      }
      // B frag (32x16 f16): lane holds column N=lane%16, K {0..7,16..23} / {8..15,24..31} like A
      v16h bf[4];
#pragma unroll
      for (int nt = 0; nt < 4; ++nt) {
        const int col = waveN + nt * 16 + lrow;
        v8h lo = *(const v8h*)&bufB[col * LDT + kk + lgrp * 8];
        v8h hi = *(const v8h*)&bufB[col * LDT + kk + 16 + lgrp * 8];
        bf[nt] = __builtin_shufflevector(lo, hi, 0, 1, 2, 3, 4, 5, 6, 7,
                                         8, 9, 10, 11, 12, 13, 14, 15);
      }
#pragma unroll
      for (int mt = 0; mt < 2; ++mt)
#pragma unroll
        for (int nt = 0; nt < 4; ++nt)
          acc[mt][nt] = __builtin_amdgcn_wmma_f32_16x16x32_f16(
              false, af[mt], false, bf[nt], (short)0, acc[mt][nt], false, false);
    }
  };

  const int KT = K / BK;
  issue_tile(0, 0u);  // prologue: tile 0 -> buffer 0

  // steady state: branch-free — issue t+1, retire t (in-order ASYNCcnt), compute t
  for (int t = 0; t < KT - 1; ++t) {
    issue_tile((t + 1) * BK, (uint32_t)(((t + 1) & 1) * TILE_H * 2));
    asm volatile("s_wait_asynccnt 0x8" ::: "memory");
    __syncthreads();  // all waves' DMA for tile t now visible
    compute_tile(&sA[(t & 1) * TILE_H], &sB[(t & 1) * TILE_H]);
    __syncthreads();  // done reading buffer (t&1) before it is re-targeted
  }

  // drain: last tile
  asm volatile("s_wait_asynccnt 0x0" ::: "memory");
  __syncthreads();
  compute_tile(&sA[((KT - 1) & 1) * TILE_H], &sB[((KT - 1) & 1) * TILE_H]);

  // epilogue: C/D layout — VGPR r holds M = r + 8*lane_group, N = lane%16
#pragma unroll
  for (int nt = 0; nt < 4; ++nt) {
    const int col = block_n + waveN + nt * 16 + lrow;
    const float bv = bias[col];
#pragma unroll
    for (int mt = 0; mt < 2; ++mt) {
      const int row0 = block_m + waveM + mt * 16 + lgrp * 8;
#pragma unroll
      for (int r = 0; r < 8; ++r)
        C[(size_t)(row0 + r) * N + col] = acc[mt][nt][r] + bv;
    }
  }
}

// ---------------------------------------------------------------------------
extern "C" void kernel_launch(void* const* d_in, const int* in_sizes, int n_in,
                              void* d_out, int out_size, void* d_ws, size_t ws_size,
                              hipStream_t stream) {
  (void)n_in; (void)out_size; (void)ws_size;
  const float* x    = (const float*)d_in[0];
  const float* w    = (const float*)d_in[1];
  const float* bias = (const float*)d_in[2];
  float* out = (float*)d_out;

  const int n_w = in_sizes[1];       // out_features * in_features
  const int N   = in_sizes[2];       // out_features (16384)
  const int K   = n_w / N;           // in_features  (4096)
  const int M   = in_sizes[0] / K;   // tokens       (8192)

  float*    d_sum = (float*)d_ws;                    // 1 float scalar
  _Float16* x_h   = (_Float16*)((char*)d_ws + 256);  // 64 MB
  _Float16* w_h   = x_h + (size_t)M * K;             // 128 MB

  hipMemsetAsync(d_sum, 0, sizeof(float), stream);
  absum_kernel<<<1024, 256, 0, stream>>>(w, d_sum, n_w / 4);
  quant_w_kernel<<<2048, 256, 0, stream>>>(w, w_h, d_sum, n_w / 4,
                                           1.0f / (float)n_w);
  convert_x_kernel<<<2048, 256, 0, stream>>>(x, x_h, in_sizes[0] / 4);

  dim3 grid(N / BN, M / BM);
  gemm_f16_wmma_async_kernel<<<grid, 256, 0, stream>>>(x_h, w_h, bias, out,
                                                       M, N, K);
}
